// GraphInputEncoder_20005957664851
// MI455X (gfx1250) — matrix operand order repaired and checked by reading, not compiled, not run on hardware
//
#include <hip/hip_runtime.h>
#include <hip/hip_bf16.h>

// =====================================================================
// GraphInputEncoder for MI455X (gfx1250).
//
// Roofline: ~145 GFLOP, dominated by K=128 GEMMs over 800k edge rows ->
// v_wmma_f32_16x16x32_bf16. Each wave computes a 32-row x DOUT strip so
// one LDS B-fragment (1KB) feeds two WMMAs (halves LDS traffic per
// FLOP, the practical WMMA-rate limiter on a WGP). W is staged into LDS
// with the CDNA5 async copy engine (global_load_async_to_lds_b128 +
// s_wait_asynccnt), then transposed once in-LDS to [col][k] so B
// fragments are two ds_load_b128 each. Attention softmax/scatter and LN
// stay fp32 (native global_atomic_add_f32, CAS float-max). bf16 storage
// halves the dominant edge streams (205MB each) vs the 23.3 TB/s roof.
// =====================================================================

#define NNODES 50000LL
#define NEDGES 800000LL
#define NP1    50001LL   // nodes + graph token
#define HID    128

typedef __bf16 v16bf __attribute__((ext_vector_type(16)));
typedef float  v8f   __attribute__((ext_vector_type(8)));

__device__ __forceinline__ float gelu_exact(float x) {
    return 0.5f * x * (1.0f + erff(x * 0.70710678118654752f));
}

__device__ __forceinline__ void atomicMaxF(float* addr, float val) {
    unsigned int* ua = (unsigned int*)addr;
    unsigned int old = *ua;
    while (__uint_as_float(old) < val) {
        unsigned int assumed = old;
        old = atomicCAS(ua, assumed, __float_as_uint(val));
        if (old == assumed) break;
    }
}

// ---------------------------------------------------------------------
// fp32 -> bf16 weight conversion
// ---------------------------------------------------------------------
__global__ void cvt_bf16_kernel(const float* __restrict__ w,
                                __hip_bfloat16* __restrict__ o, int n) {
    int g = blockIdx.x * blockDim.x + threadIdx.x;
    if (g < n) o[g] = __float2bfloat16(w[g]);
}

// ---------------------------------------------------------------------
// fill
// ---------------------------------------------------------------------
__global__ void fill_f32_kernel(float* __restrict__ p, float v, long long n) {
    long long g = (long long)blockIdx.x * blockDim.x + threadIdx.x;
    if (g < n) p[g] = v;
}

// ---------------------------------------------------------------------
// WMMA GEMM: C[M,DOUT] = A[M,128](bf16) @ W[128,DOUT](bf16) (+bias)
// 256 threads = 8 waves; wave w owns rows [blk*256 + 32w, +32)
// (two 16-row WMMA tiles sharing each B fragment).
// W async-staged row-major into LDS, then transposed to ldsW[c*128+k].
// ---------------------------------------------------------------------
template <int DOUT, bool BIAS, bool OUTF, bool OUTB>
__global__ __launch_bounds__(256) void gemm_bf16_wmma(
    const __hip_bfloat16* __restrict__ A, const __hip_bfloat16* __restrict__ W,
    const float* __restrict__ bias, float* __restrict__ outF,
    __hip_bfloat16* __restrict__ outB, long long M) {
    constexpr int NT = DOUT / 16;  // 16-col tiles per wave
    __shared__ unsigned short ldsRaw[128 * DOUT];  // row-major [k][c]
    __shared__ unsigned short ldsW[128 * DOUT];    // transposed [c][k]

    const int tid = threadIdx.x;

    // --- async copy W (row-major) into LDS: 16B per lane per op ---
    {
        const unsigned long long wg = (unsigned long long)W;
        for (int idx = tid * 8; idx < 128 * DOUT; idx += 256 * 8) {
            unsigned lofs = (unsigned)(unsigned long long)&ldsRaw[idx];
            unsigned long long ga = wg + (unsigned long long)idx * 2;
            asm volatile("global_load_async_to_lds_b128 %0, %1, off"
                         :: "v"(lofs), "v"(ga) : "memory");
        }
        asm volatile("s_wait_asynccnt 0x0" ::: "memory");
        __syncthreads();
        // one-time in-LDS transpose to [c][k]
        for (int idx = tid; idx < 128 * DOUT; idx += 256) {
            int k = idx / DOUT, c = idx % DOUT;
            ldsW[c * 128 + k] = ldsRaw[idx];
        }
        __syncthreads();
    }

    const int wave = tid >> 5;
    const int lane = tid & 31;
    const long long rowBase = (long long)blockIdx.x * 256 + wave * 32;
    long long r0 = rowBase + (lane & 15);
    long long r1 = rowBase + 16 + (lane & 15);
    if (r0 >= M) r0 = M - 1;
    if (r1 >= M) r1 = M - 1;
    const unsigned short* Ab = (const unsigned short*)A;
    const unsigned short* ar0 = Ab + r0 * 128;
    const unsigned short* ar1 = Ab + r1 * 128;

    // CDNA5 prefetch of the next row block (global_prefetch_b8)
    if (rowBase + 256 + (lane & 15) < M)
        __builtin_prefetch((const void*)(Ab + (rowBase + 256 + (lane & 15)) * 128), 0, 1);

    union AFrag { v16bf v; unsigned int u[8]; };
    union BFrag { v16bf v; uint4 q[2]; };

    v8f acc0[NT] = {};
    v8f acc1[NT] = {};
    const int h8 = (lane >> 4) << 3;    // A: +8 K for upper lane half
    const int kb16 = (lane >> 4) << 3;  // B: +8 K for upper lane half (K {0..7|16..23}+8h)
    const int colb = lane & 15;

#pragma unroll
    for (int ks = 0; ks < 4; ++ks) {
        const int kk = ks * 32;
        AFrag af0, af1;
        // 16-bit A 16x32 layout: lanes 0-15 row M=lane, K pairs
        // {0..7 | 16..23}; lanes 16-31 same row, K shifted by +8.
#pragma unroll
        for (int p = 0; p < 4; ++p) {
            af0.u[p]     = *(const unsigned int*)(ar0 + kk + 2 * p + h8);
            af0.u[4 + p] = *(const unsigned int*)(ar0 + kk + 16 + 2 * p + h8);
            af1.u[p]     = *(const unsigned int*)(ar1 + kk + 2 * p + h8);
            af1.u[4 + p] = *(const unsigned int*)(ar1 + kk + 16 + 2 * p + h8);
        }

#pragma unroll
        for (int t = 0; t < NT; ++t) {
            BFrag bf;
            const int col = t * 16 + colb;
            const uint4* lp = (const uint4*)&ldsW[col * 128 + kk + kb16];
            bf.q[0] = lp[0];
            bf.q[1] = lp[2];
            acc0[t] = __builtin_amdgcn_wmma_f32_16x16x32_bf16(
                false, af0.v, false, bf.v, (short)0, acc0[t], false, false);
            acc1[t] = __builtin_amdgcn_wmma_f32_16x16x32_bf16(
                false, af1.v, false, bf.v, (short)0, acc1[t], false, false);
        }
    }

    // C/D layout: VGPR v -> row v (lanes 0-15) / v+8 (lanes 16-31); col = lane&15
    const int mofs = (lane >> 4) << 3;
#pragma unroll
    for (int t = 0; t < NT; ++t) {
        const int col = t * 16 + colb;
        float bv = 0.0f;
        if constexpr (BIAS) bv = bias[col];
#pragma unroll
        for (int vv = 0; vv < 8; ++vv) {
            long long ra = rowBase + vv + mofs;
            long long rb = ra + 16;
            if (ra < M) {
                float val = acc0[t][vv] + bv;
                if constexpr (OUTF) outF[ra * DOUT + col] = val;
                if constexpr (OUTB) outB[ra * DOUT + col] = __float2bfloat16(val);
            }
            if (rb < M) {
                float val = acc1[t][vv] + bv;
                if constexpr (OUTF) outF[rb * DOUT + col] = val;
                if constexpr (OUTB) outB[rb * DOUT + col] = __float2bfloat16(val);
            }
        }
    }
}

// ---------------------------------------------------------------------
// graph token: t = W2 @ LN(gelu(W1 @ gf + b1)) + b2  -> x row 0
// ---------------------------------------------------------------------
__global__ __launch_bounds__(128) void graph_token_kernel(
    const float* __restrict__ gf, const float* __restrict__ W1,
    const float* __restrict__ b1, const float* __restrict__ lg,
    const float* __restrict__ lb, const float* __restrict__ W2,
    const float* __restrict__ b2, float* __restrict__ xF,
    __hip_bfloat16* __restrict__ xB) {
    __shared__ float t[128];
    __shared__ float red[128];
    const int c = threadIdx.x;
    float v = b1[c];
#pragma unroll
    for (int k = 0; k < 14; ++k) v += gf[k] * W1[k * 128 + c];
    v = gelu_exact(v);
    red[c] = v; __syncthreads();
    for (int s = 64; s > 0; s >>= 1) { if (c < s) red[c] += red[c + s]; __syncthreads(); }
    float mean = red[0] * (1.0f / 128.0f); __syncthreads();
    float d = v - mean;
    red[c] = d * d; __syncthreads();
    for (int s = 64; s > 0; s >>= 1) { if (c < s) red[c] += red[c + s]; __syncthreads(); }
    float var = red[0] * (1.0f / 128.0f);
    float y = d * rsqrtf(var + 1e-5f) * lg[c] + lb[c];
    t[c] = y; __syncthreads();
    float o = b2[c];
    for (int k = 0; k < 128; ++k) o += t[k] * W2[k * 128 + c];
    xF[c] = o;
    xB[c] = __float2bfloat16(o);
}

// ---------------------------------------------------------------------
// node projection: x[1+n] = nf[n] @ W[9,128] + b
// ---------------------------------------------------------------------
__global__ __launch_bounds__(256) void node_proj_kernel(
    const float* __restrict__ nf, const float* __restrict__ W,
    const float* __restrict__ b, float* __restrict__ xF,
    __hip_bfloat16* __restrict__ xB) {
    long long g = (long long)blockIdx.x * blockDim.x + threadIdx.x;
    if (g >= NNODES * 128) return;
    long long n = g >> 7;
    int c = (int)(g & 127);
    const float* r = nf + n * 9;
    float v = b[c];
#pragma unroll
    for (int k = 0; k < 9; ++k) v += r[k] * W[k * 128 + c];
    long long row = n + 1;
    xF[row * 128 + c] = v;
    xB[row * 128 + c] = __float2bfloat16(v);
}

// ---------------------------------------------------------------------
// edge MLP l1: h = LN(gelu(ef @ W1[3,128] + b1)) -> bf16 scratch
// ---------------------------------------------------------------------
__global__ __launch_bounds__(128) void edge_mlp1_kernel(
    const float* __restrict__ ef, const float* __restrict__ W1,
    const float* __restrict__ b1, const float* __restrict__ lg,
    const float* __restrict__ lb, __hip_bfloat16* __restrict__ out) {
    __shared__ float red[128];
    const long long e = blockIdx.x;
    const int c = threadIdx.x;
    const float* er = ef + e * 3;
    float v = b1[c] + er[0] * W1[c] + er[1] * W1[128 + c] + er[2] * W1[256 + c];
    v = gelu_exact(v);
    red[c] = v; __syncthreads();
    for (int s = 64; s > 0; s >>= 1) { if (c < s) red[c] += red[c + s]; __syncthreads(); }
    float mean = red[0] * (1.0f / 128.0f); __syncthreads();
    float d = v - mean;
    red[c] = d * d; __syncthreads();
    for (int s = 64; s > 0; s >>= 1) { if (c < s) red[c] += red[c + s]; __syncthreads(); }
    float var = red[0] * (1.0f / 128.0f);
    float y = d * rsqrtf(var + 1e-5f) * lg[c] + lb[c];
    out[e * 128 + c] = __float2bfloat16(y);
}

// ---------------------------------------------------------------------
// attention pass 1: alpha = q[dst].(k[src]+ep) / sqrt(C); segment max
// ---------------------------------------------------------------------
template <int C>
__global__ __launch_bounds__(256) void attn_alpha_kernel(
    const int* __restrict__ ei, const float* __restrict__ q,
    const float* __restrict__ k, const __hip_bfloat16* __restrict__ ep,
    float* __restrict__ alpha, float* __restrict__ amax) {
    long long g = (long long)blockIdx.x * blockDim.x + threadIdx.x;
    if (g >= NEDGES * 4) return;
    long long e = g >> 2;
    int h = (int)(g & 3);
    long long src = (long long)ei[e] + 1;
    long long dst = (long long)ei[NEDGES + e] + 1;
    constexpr int DO = 4 * C;
    const float* qr = q + dst * DO + h * C;
    const float* kr = k + src * DO + h * C;
    const __hip_bfloat16* er = ep + e * DO + h * C;
    float s = 0.0f;
#pragma unroll
    for (int c = 0; c < C; ++c) s += qr[c] * (kr[c] + __bfloat162float(er[c]));
    s *= (C == 32) ? 0.17677669529663687f : 0.25f;  // 1/sqrt(C)
    alpha[g] = s;
    atomicMaxF(&amax[dst * 4 + h], s);
}

// ---------------------------------------------------------------------
// attention pass 2: ex = exp(alpha - amax[dst]); segment sum
// ---------------------------------------------------------------------
__global__ __launch_bounds__(256) void attn_ex_kernel(
    const int* __restrict__ ei, const float* __restrict__ alpha,
    const float* __restrict__ amax, float* __restrict__ exb,
    float* __restrict__ den) {
    long long g = (long long)blockIdx.x * blockDim.x + threadIdx.x;
    if (g >= NEDGES * 4) return;
    long long e = g >> 2;
    int h = (int)(g & 3);
    long long dst = (long long)ei[NEDGES + e] + 1;
    float m = amax[dst * 4 + h];
    if (!(m > -3.0e38f)) m = 0.0f;  // -inf (untouched segment) -> 0
    float ev = expf(alpha[g] - m);
    exb[g] = ev;
    atomicAdd(&den[dst * 4 + h], ev);
}

// ---------------------------------------------------------------------
// attention pass 3: agg[dst] += (v[src]+ep) * ex/den[dst]
// ---------------------------------------------------------------------
template <int C>
__global__ __launch_bounds__(256) void attn_msg_kernel(
    const int* __restrict__ ei, const float* __restrict__ v,
    const __hip_bfloat16* __restrict__ ep, const float* __restrict__ exb,
    const float* __restrict__ den, float* __restrict__ agg) {
    long long g = (long long)blockIdx.x * blockDim.x + threadIdx.x;
    if (g >= NEDGES * 4) return;
    long long e = g >> 2;
    int h = (int)(g & 3);
    long long src = (long long)ei[e] + 1;
    long long dst = (long long)ei[NEDGES + e] + 1;
    constexpr int DO = 4 * C;
    float d = den[dst * 4 + h];
    if (!(d > 0.0f)) d = 1.0f;
    float w = exb[g] / d;
    const float* vr = v + src * DO + h * C;
    const __hip_bfloat16* er = ep + e * DO + h * C;
    float* ar = agg + dst * DO + h * C;
#pragma unroll
    for (int c = 0; c < C; ++c)
        atomicAdd(&ar[c], (vr[c] + __bfloat162float(er[c])) * w);
}

// ---------------------------------------------------------------------
// post: h = agg + skip (+x); LN; (gelu); write next x (or final output)
// ---------------------------------------------------------------------
template <int DO, bool RES, bool GELU, bool LAST>
__global__ void post_ln_kernel(const float* __restrict__ agg,
                               const float* __restrict__ skip,
                               const float* __restrict__ xin,
                               const float* __restrict__ g,
                               const float* __restrict__ b,
                               float* __restrict__ xF,
                               __hip_bfloat16* __restrict__ xB,
                               float* __restrict__ out) {
    __shared__ float red[DO];
    const long long row = blockIdx.x;
    const int c = threadIdx.x;
    float h = agg[row * DO + c] + skip[row * DO + c];
    if constexpr (RES) h += xin[row * 128 + c];
    red[c] = h; __syncthreads();
    for (int s = DO / 2; s > 0; s >>= 1) { if (c < s) red[c] += red[c + s]; __syncthreads(); }
    float mean = red[0] * (1.0f / DO); __syncthreads();
    float d = h - mean;
    red[c] = d * d; __syncthreads();
    for (int s = DO / 2; s > 0; s >>= 1) { if (c < s) red[c] += red[c + s]; __syncthreads(); }
    float var = red[0] * (1.0f / DO);
    float y = d * rsqrtf(var + 1e-5f) * g[c] + b[c];
    if constexpr (GELU) y = gelu_exact(y);
    if constexpr (LAST) {
        out[row * DO + c] = y;
    } else {
        xF[row * 128 + c] = y;
        xB[row * 128 + c] = __float2bfloat16(y);
    }
}

// =====================================================================
// host-side orchestration
// =====================================================================
extern "C" void kernel_launch(void* const* d_in, const int* in_sizes, int n_in,
                              void* d_out, int out_size, void* d_ws, size_t ws_size,
                              hipStream_t stream) {
    // input order: dict insertion order, then params flattened jax-style
    // (sorted dict keys, lists in order):
    //  0 graph_features[14]  1 node_features[N,9]  2 edge_features[E,3]
    //  3 edge_index[2,E]
    //  4 em.l1.W  5 em.l1.b  6 em.l2.W  7 em.l2.b  8 em.ln_b  9 em.ln_g
    // 10 gm.l1.W 11 gm.l1.b 12 gm.l2.W 13 gm.l2.b 14 gm.ln_b 15 gm.ln_g
    // layer i (base 16+11i): +0 e.W +1 k.W +2 k.b +3 ln_b +4 ln_g
    //                        +5 q.W +6 q.b +7 s.W +8 s.b +9 v.W +10 v.b
    // 60 np.W  61 np.b
    const float* gf = (const float*)d_in[0];
    const float* nf = (const float*)d_in[1];
    const float* ef = (const float*)d_in[2];
    const int*   ei = (const int*)d_in[3];

    char* ws = (char*)d_ws;
    size_t off = 0;
    auto take = [&](size_t bytes) {
        size_t o = off;
        off = (off + bytes + 255) & ~(size_t)255;
        return o;
    };
    float*          xF    = (float*)(ws + take(NP1 * 128 * 4));
    __hip_bfloat16* xB    = (__hip_bfloat16*)(ws + take(NP1 * 128 * 2));
    __hip_bfloat16* eB    = (__hip_bfloat16*)(ws + take(NEDGES * 128 * 2));
    __hip_bfloat16* sE    = (__hip_bfloat16*)(ws + take(NEDGES * 128 * 2)); // l1 hidden, then ep
    float*          qF    = (float*)(ws + take(NP1 * 128 * 4));
    float*          kF    = (float*)(ws + take(NP1 * 128 * 4));
    float*          vF    = (float*)(ws + take(NP1 * 128 * 4));
    float*          sF    = (float*)(ws + take(NP1 * 128 * 4));
    float*          alpha = (float*)(ws + take(NEDGES * 4 * 4));
    float*          exb   = (float*)(ws + take(NEDGES * 4 * 4));
    float*          amax  = (float*)(ws + take(NP1 * 4 * 4));
    float*          den   = (float*)(ws + take(NP1 * 4 * 4));
    float*          agg   = (float*)(ws + take(NP1 * 128 * 4));
    __hip_bfloat16* wb    = (__hip_bfloat16*)(ws + take(303104 * 2));

    // ---- convert GEMM weights to bf16 ----
    size_t wpos = 0;
    auto cvt = [&](const void* src, int n) -> __hip_bfloat16* {
        __hip_bfloat16* dst = wb + wpos;
        wpos += (size_t)n;
        cvt_bf16_kernel<<<(n + 255) / 256, 256, 0, stream>>>((const float*)src, dst, n);
        return dst;
    };
    __hip_bfloat16* w_el2 = cvt(d_in[6], 128 * 128);
    __hip_bfloat16 *w_e[4], *w_k[4], *w_q[4], *w_s[4], *w_v[4];
    for (int i = 0; i < 4; ++i) {
        const int dO = (i < 3) ? 128 : 64;
        const int b  = 16 + 11 * i;
        w_e[i] = cvt(d_in[b + 0], 128 * dO);
        w_k[i] = cvt(d_in[b + 1], 128 * dO);
        w_q[i] = cvt(d_in[b + 5], 128 * dO);
        w_s[i] = cvt(d_in[b + 7], 128 * dO);
        w_v[i] = cvt(d_in[b + 9], 128 * dO);
    }

    // ---- front end ----
    graph_token_kernel<<<1, 128, 0, stream>>>(
        gf, (const float*)d_in[10], (const float*)d_in[11],
        (const float*)d_in[15], (const float*)d_in[14],
        (const float*)d_in[12], (const float*)d_in[13], xF, xB);
    node_proj_kernel<<<(int)((NNODES * 128 + 255) / 256), 256, 0, stream>>>(
        nf, (const float*)d_in[60], (const float*)d_in[61], xF, xB);
    edge_mlp1_kernel<<<(int)NEDGES, 128, 0, stream>>>(
        ef, (const float*)d_in[4], (const float*)d_in[5],
        (const float*)d_in[9], (const float*)d_in[8], sE);
    // edge MLP l2 (bf16 WMMA GEMM): eB = sE @ W_l2 + b
    const int egrid = (int)((NEDGES + 255) / 256);
    gemm_bf16_wmma<128, true, false, true><<<egrid, 256, 0, stream>>>(
        sE, w_el2, (const float*)d_in[7], nullptr, eB, NEDGES);

    // ---- transformer layers ----
    const int ngrid = (int)((NP1 + 255) / 256);
    const int agrid = (int)((NEDGES * 4 + 255) / 256);
    for (int i = 0; i < 4; ++i) {
        const int b = 16 + 11 * i;
        const float* kb  = (const float*)d_in[b + 2];
        const float* lnb = (const float*)d_in[b + 3];
        const float* lng = (const float*)d_in[b + 4];
        const float* qb  = (const float*)d_in[b + 6];
        const float* sb  = (const float*)d_in[b + 8];
        const float* vb  = (const float*)d_in[b + 10];
        if (i < 3) {
            gemm_bf16_wmma<128, true, true, false><<<ngrid, 256, 0, stream>>>(
                xB, w_q[i], qb, qF, nullptr, NP1);
            gemm_bf16_wmma<128, true, true, false><<<ngrid, 256, 0, stream>>>(
                xB, w_k[i], kb, kF, nullptr, NP1);
            gemm_bf16_wmma<128, true, true, false><<<ngrid, 256, 0, stream>>>(
                xB, w_v[i], vb, vF, nullptr, NP1);
            gemm_bf16_wmma<128, true, true, false><<<ngrid, 256, 0, stream>>>(
                xB, w_s[i], sb, sF, nullptr, NP1);
            gemm_bf16_wmma<128, false, false, true><<<egrid, 256, 0, stream>>>(
                eB, w_e[i], nullptr, nullptr, sE, NEDGES);  // ep
            fill_f32_kernel<<<(int)((NP1 * 4 + 255) / 256), 256, 0, stream>>>(
                amax, -__builtin_huge_valf(), NP1 * 4);
            fill_f32_kernel<<<(int)((NP1 * 4 + 255) / 256), 256, 0, stream>>>(
                den, 0.0f, NP1 * 4);
            fill_f32_kernel<<<(int)((NP1 * 128 + 255) / 256), 256, 0, stream>>>(
                agg, 0.0f, NP1 * 128);
            attn_alpha_kernel<32><<<agrid, 256, 0, stream>>>(ei, qF, kF, sE, alpha, amax);
            attn_ex_kernel<<<agrid, 256, 0, stream>>>(ei, alpha, amax, exb, den);
            attn_msg_kernel<32><<<agrid, 256, 0, stream>>>(ei, vF, sE, exb, den, agg);
            post_ln_kernel<128, true, true, false><<<(int)NP1, 128, 0, stream>>>(
                agg, sF, xF, lng, lnb, xF, xB, nullptr);
        } else {
            gemm_bf16_wmma<64, true, true, false><<<ngrid, 256, 0, stream>>>(
                xB, w_q[i], qb, qF, nullptr, NP1);
            gemm_bf16_wmma<64, true, true, false><<<ngrid, 256, 0, stream>>>(
                xB, w_k[i], kb, kF, nullptr, NP1);
            gemm_bf16_wmma<64, true, true, false><<<ngrid, 256, 0, stream>>>(
                xB, w_v[i], vb, vF, nullptr, NP1);
            gemm_bf16_wmma<64, true, true, false><<<ngrid, 256, 0, stream>>>(
                xB, w_s[i], sb, sF, nullptr, NP1);
            gemm_bf16_wmma<64, false, false, true><<<egrid, 256, 0, stream>>>(
                eB, w_e[i], nullptr, nullptr, sE, NEDGES);
            fill_f32_kernel<<<(int)((NP1 * 4 + 255) / 256), 256, 0, stream>>>(
                amax, -__builtin_huge_valf(), NP1 * 4);
            fill_f32_kernel<<<(int)((NP1 * 4 + 255) / 256), 256, 0, stream>>>(
                den, 0.0f, NP1 * 4);
            fill_f32_kernel<<<(int)((NP1 * 64 + 255) / 256), 256, 0, stream>>>(
                agg, 0.0f, NP1 * 64);
            attn_alpha_kernel<16><<<agrid, 256, 0, stream>>>(ei, qF, kF, sE, alpha, amax);
            attn_ex_kernel<<<agrid, 256, 0, stream>>>(ei, alpha, amax, exb, den);
            attn_msg_kernel<16><<<agrid, 256, 0, stream>>>(ei, vF, sE, exb, den, agg);
            post_ln_kernel<64, false, false, true><<<(int)NP1, 64, 0, stream>>>(
                agg, sF, nullptr, lng, lnb, nullptr, nullptr, (float*)d_out);
        }
    }
    (void)in_sizes; (void)n_in; (void)out_size; (void)ws_size;
}
